// nll_l_cvm_76132590289220
// MI455X (gfx1250) — matrix operand order repaired and checked by reading, not compiled, not run on hardware
//
#include <hip/hip_runtime.h>

typedef __attribute__((ext_vector_type(16))) _Float16 v16h;
typedef __attribute__((ext_vector_type(8)))  float    v8f;
typedef __attribute__((ext_vector_type(4)))  float    f32x4;

#define DDIM 64
#define PDIM 32
#define TILE_ROWS 16
#define BLOCKS 2048
#define TPB 256
#define WPB (TPB / 32)

#define LOG2E 1.4426950408889634f
#define LN2   0.6931471805599453f

// etaT = (W*log2e) @ V^T via WMMA, d on the M axis so logsumexp over d is a
// per-lane register reduction + one cross-half exchange. Accumulators are in
// log2 units, so exp/log use the native base-2 ops with no per-element scale;
// one final *ln2 restores natural-log units for both q1 and q2.
__global__ __launch_bounds__(TPB) void nll_fused_kernel(
    const float* __restrict__ C,
    const float* __restrict__ V,
    const float* __restrict__ m,
    const float* __restrict__ W,
    float* __restrict__ ws,
    int n)
{
    const int lane   = threadIdx.x & 31;
    const int laneLo = lane & 15;
    const int half   = lane >> 4;          // 0: lanes 0-15, 1: lanes 16-31
    const int waveId = threadIdx.x >> 5;
    const int wave   = blockIdx.x * WPB + waveId;
    const int nWaves = gridDim.x * WPB;
    const int nTiles = n / TILE_ROWS;      // 62500

    // ---- A tiles = W rows * log2e (d is M), loop-invariant: once per wave.
    // 16-bit A 16x32 layout: lane holds row laneLo; elem e -> K = e+(e>=8?8:0)+8*half.
    v16h a[4];
    #pragma unroll
    for (int t = 0; t < 4; ++t) {
        const float* wp = W + (t * 16 + laneLo) * PDIM + 8 * half;
        f32x4 w0 = *(const f32x4*)(wp + 0);
        f32x4 w1 = *(const f32x4*)(wp + 4);
        f32x4 w2 = *(const f32x4*)(wp + 16);
        f32x4 w3 = *(const f32x4*)(wp + 20);
        #pragma unroll
        for (int e = 0; e < 4; ++e) {
            a[t][e]      = (_Float16)(w0[e] * LOG2E);
            a[t][4 + e]  = (_Float16)(w1[e] * LOG2E);
            a[t][8 + e]  = (_Float16)(w2[e] * LOG2E);
            a[t][12 + e] = (_Float16)(w3[e] * LOG2E);
        }
    }

    float q1part = 0.f, q2part = 0.f;   // both in log2-scaled units

    for (int tile = wave; tile < nTiles; tile += nWaves) {
        const int base = tile * TILE_ROWS;
        const int nIdx = base + laneLo;    // this lane's sample/column index

        // ---- B tile = V^T (32 K x 16 cols). 16-bit B 32x16 layout: lane holds
        // col laneLo, elem e -> K = e+(e>=8?8:0)+8*half (matches A), row nIdx of V.
        const float* vp = V + (size_t)nIdx * PDIM + 8 * half;
        f32x4 v0 = __builtin_nontemporal_load((const f32x4*)(vp + 0));
        f32x4 v1 = __builtin_nontemporal_load((const f32x4*)(vp + 4));
        f32x4 v2 = __builtin_nontemporal_load((const f32x4*)(vp + 16));
        f32x4 v3 = __builtin_nontemporal_load((const f32x4*)(vp + 20));
        v16h b;
        #pragma unroll
        for (int e = 0; e < 4; ++e) {
            b[e]      = (_Float16)v0[e];
            b[4 + e]  = (_Float16)v1[e];
            b[8 + e]  = (_Float16)v2[e];
            b[12 + e] = (_Float16)v3[e];
        }

        // ---- 4 WMMAs sharing B: etaT*log2e for 64 d-rows x 16 columns.
        v8f acc[4];
        #pragma unroll
        for (int t = 0; t < 4; ++t) {
            v8f z = {};
            acc[t] = __builtin_amdgcn_wmma_f32_16x16x32_f16(
                false, a[t], false, b, (short)0, z, false, false);
        }
        // lane holds log2e*etaT[d = t*16 + v + 8*half][nIdx] for v=0..7, t=0..3
        // (32 of the 64 d-values; partner lane (lane^16) holds the rest).

        // ---- local max over this lane's 32 d-values.
        float mx = acc[0][0];
        #pragma unroll
        for (int t = 0; t < 4; ++t)
            #pragma unroll
            for (int v = 0; v < 8; ++v)
                mx = fmaxf(mx, acc[t][v]);

        // ---- q2 partial + local exp2-sum. C values for this lane are 4
        // contiguous 8-float chunks of row nIdx (offset t*16 + 8*half).
        const float* cp = C + (size_t)nIdx * DDIM + 8 * half;
        float s = 0.f;
        #pragma unroll
        for (int t = 0; t < 4; ++t) {
            f32x4 c0 = __builtin_nontemporal_load((const f32x4*)(cp + t * 16));
            f32x4 c1 = __builtin_nontemporal_load((const f32x4*)(cp + t * 16 + 4));
            #pragma unroll
            for (int v = 0; v < 4; ++v) {
                q2part = fmaf(c0[v], acc[t][v], q2part);
                q2part = fmaf(c1[v], acc[t][v + 4], q2part);
                s += __builtin_amdgcn_exp2f(acc[t][v] - mx);       // v_exp_f32
                s += __builtin_amdgcn_exp2f(acc[t][v + 4] - mx);
            }
        }

        // ---- merge with partner half (one exchange pair per tile).
        float mxP = __shfl_xor(mx, 16, 32);
        float sP  = __shfl_xor(s, 16, 32);
        float M   = fmaxf(mx, mxP);
        float st  = s * __builtin_amdgcn_exp2f(mx - M)
                  + sP * __builtin_amdgcn_exp2f(mxP - M);
        float lse2 = M + __builtin_amdgcn_logf(st);                // v_log_f32 = log2
        if (half == 0)
            q1part += m[nIdx] * lse2;
    }

    // ---- deterministic block reduction of ln2*(q1 - q2).
    float part = (q1part - q2part) * LN2;
    #pragma unroll
    for (int d = 1; d < 32; d <<= 1)
        part += __shfl_xor(part, d, 32);

    __shared__ float sred[WPB];
    if (lane == 0) sred[waveId] = part;
    __syncthreads();
    if (threadIdx.x == 0) {
        float tot = 0.f;
        #pragma unroll
        for (int i = 0; i < WPB; ++i) tot += sred[i];
        ws[blockIdx.x] = tot;
    }
}

__global__ __launch_bounds__(256) void nll_finalize_kernel(
    const float* __restrict__ ws, float* __restrict__ out, int nparts, float invN)
{
    __shared__ float sred[256];
    float s = 0.f;
    for (int i = threadIdx.x; i < nparts; i += 256) s += ws[i];
    sred[threadIdx.x] = s;
    __syncthreads();
    for (int stride = 128; stride > 0; stride >>= 1) {
        if (threadIdx.x < stride) sred[threadIdx.x] += sred[threadIdx.x + stride];
        __syncthreads();
    }
    if (threadIdx.x == 0) out[0] = sred[0] * invN;
}

extern "C" void kernel_launch(void* const* d_in, const int* in_sizes, int n_in,
                              void* d_out, int out_size, void* d_ws, size_t ws_size,
                              hipStream_t stream) {
    const float* C = (const float*)d_in[0];   // [N, 64]
    const float* V = (const float*)d_in[1];   // [N, 32]
    const float* m = (const float*)d_in[2];   // [N]
    const float* W = (const float*)d_in[3];   // [64, 32]
    const int n = in_sizes[2];                // N = 1,000,000 (divisible by 16)
    float* ws = (float*)d_ws;                 // BLOCKS partials (8 KB)

    nll_fused_kernel<<<BLOCKS, TPB, 0, stream>>>(C, V, m, W, ws, n);
    nll_finalize_kernel<<<1, 256, 0, stream>>>(ws, (float*)d_out, BLOCKS,
                                               1.0f / (float)n);
}
